// NetVLAD_56478819942785
// MI455X (gfx1250) — matrix operand order repaired and checked by reading, not compiled, not run on hardware
//
#include <hip/hip_runtime.h>
#include <hip/hip_bf16.h>
#include <type_traits>

typedef __attribute__((ext_vector_type(16))) __bf16       v16bf;
typedef __attribute__((ext_vector_type(8)))  float        v8f;
typedef __attribute__((ext_vector_type(4)))  unsigned int v4u;
typedef __attribute__((ext_vector_type(4)))  unsigned int u32x4;
typedef __attribute__((ext_vector_type(8)))  int          i32x8;
typedef __attribute__((ext_vector_type(4)))  int          i32x4;

#define BSZ   16
#define DIMC  1024
#define NUMP  4096
#define KCLU  32
#define PROJD 128
#define NTOT  (BSZ * NUMP)   /* 65536 */

__device__ inline __bf16 to_bf16(float  v) { return (__bf16)v; }
__device__ inline __bf16 to_bf16(__bf16 v) { return v; }

union Frag { v16bf v; v4u q[2]; };

// ---------------------------------------------------------------------------
// Generic WMMA GEMM:  C[m][n] = sum_k A[m][k] * B[k][n]  (+ epilogue)
//   A: bf16 row-major [M][K], lda (elements); staged to LDS by the TDM
//      (tensor_load_to_lds), tile = MTILE x 32 bf16, 64B rows + 16B LDS pad.
//   B: BT=false -> B[k][n] = Bp[k*ldb + n];  BT=true -> B[k][n] = Bp[n*ldb + k]
//      bcols>0: tile's batch index = n0/bcols, Bp += b*bsplit, col -= b*bcols
//   EPI==0: +bias[m];  EPI==1: -asum[z*32+m]*cent[m*DIMC+n]  (VLAD)
// ---------------------------------------------------------------------------
template<int MTILE, int NTILE, int WM, int WN, typename TB, bool BT, int EPI>
__global__ __launch_bounds__(256)
void gemm_wmma(const __bf16* __restrict__ A, long lda, long abstride,
               const TB* __restrict__ B, long ldb, long bbstride,
               int bcols, long bsplit,
               float* __restrict__ C, long ldc, long cbstride,
               const float* __restrict__ bias,
               const float* __restrict__ asum,
               const float* __restrict__ cent,
               int K)
{
    static_assert(WM * WN == 8, "8 waves / 256 threads");
    constexpr int MSUB = MTILE / (16 * WM);
    constexpr int NSUB = NTILE / (16 * WN);
    constexpr int AST  = 40;                      // padded row (bf16 units), 80B
    __shared__ __bf16 sA[MTILE * AST];
    __shared__ __bf16 sB[NTILE * AST];

    const int tid  = threadIdx.x;
    const int lane = tid & 31;
    const int wid  = tid >> 5;
    const int wm   = wid / WN;
    const int wn   = wid % WN;
    const int m0   = blockIdx.y * MTILE;
    const int n0   = blockIdx.x * NTILE;
    const int z    = blockIdx.z;
    const int r    = lane & 15;
    const int hf   = lane >> 4;

    const __bf16* Ab = A + (long)z * abstride;
    const TB*     Bb = B + (long)z * bbstride;
    long ncol0 = n0;
    if (bcols > 0) {
        int bidx = n0 / bcols;
        Bb += (long)bidx * bsplit;
        ncol0 = n0 - (long)bidx * bcols;
    }

    v8f acc[MSUB][NSUB];
    #pragma unroll
    for (int ms = 0; ms < MSUB; ++ms)
        #pragma unroll
        for (int ns = 0; ns < NSUB; ++ns)
            #pragma unroll
            for (int i = 0; i < 8; ++i) acc[ms][ns][i] = 0.0f;

    for (int kt = 0; kt < K; kt += 32) {
        if (kt + 32 < K) __builtin_prefetch(Ab + (long)m0 * lda + kt + 32, 0, 1);

        // ---- stage A tile (MTILE x 32 bf16) via Tensor Data Mover ----------
        if (wid == 0) {
            const __bf16* gptr = Ab + (long)m0 * lda + kt;
            unsigned long long ga = (unsigned long long)(uintptr_t)gptr;
            u32x4 g0;
            g0[0] = 1u;                                          // count=1, user desc
            g0[1] = (unsigned)(uintptr_t)(void*)sA;              // lds_addr (bytes)
            g0[2] = (unsigned)(ga & 0xffffffffull);              // global_addr lo
            g0[3] = (unsigned)((ga >> 32) & 0x1ffffffull)        // global_addr hi
                    | (2u << 30);                                // type = 2 (image)
            i32x8 g1;
            g1[0] = (1 << 16)      /* data_size = 2B      */
                  | (1 << 20)      /* pad_enable          */
                  | (3 << 22)      /* pad_interval: 16 DW */
                  | (3 << 25);     /* pad_amount:   4 DW  */
            g1[1] = (int)(32u << 16);                 // tensor_dim0 = 32
            g1[2] = (int)((unsigned)MTILE << 16);     // tensor_dim1 = MTILE
            g1[3] = (int)(32u << 16);                 // tile_dim0 = 32
            g1[4] = MTILE;                            // tile_dim1 = MTILE
            g1[5] = (int)(unsigned)lda;               // tensor_dim0_stride
            g1[6] = 0;
            g1[7] = 0;
            i32x4 gz;
            gz[0] = 0; gz[1] = 0; gz[2] = 0; gz[3] = 0;
            i32x8 gz8;
            gz8[0] = 0; gz8[1] = 0; gz8[2] = 0; gz8[3] = 0;
            gz8[4] = 0; gz8[5] = 0; gz8[6] = 0; gz8[7] = 0;
            __builtin_amdgcn_tensor_load_to_lds(g0, g1, gz, gz, gz8, 0);
            __builtin_amdgcn_s_wait_tensorcnt(0);
        }

        // ---- stage B tile transposed in LDS: sB[n][k] ----------------------
        if (BT) {
            #pragma unroll
            for (int idx = tid; idx < NTILE * 32; idx += 256) {
                int n = idx >> 5, k = idx & 31;
                sB[n * AST + k] = to_bf16(Bb[(ncol0 + n) * ldb + (kt + k)]);
            }
        } else {
            #pragma unroll
            for (int idx = tid; idx < NTILE * 32; idx += 256) {
                int n = idx % NTILE, k = idx / NTILE;
                sB[n * AST + k] = to_bf16(Bb[(long)(kt + k) * ldb + ncol0 + n]);
            }
        }
        __syncthreads();

        Frag afr[MSUB], bfr[NSUB];
        #pragma unroll
        for (int ms = 0; ms < MSUB; ++ms) {
            int row = (wm * MSUB + ms) * 16 + r;
            afr[ms].q[0] = *reinterpret_cast<const v4u*>(sA + row * AST + 8 * hf);
            afr[ms].q[1] = *reinterpret_cast<const v4u*>(sA + row * AST + 16 + 8 * hf);
        }
        #pragma unroll
        for (int ns = 0; ns < NSUB; ++ns) {
            int col = (wn * NSUB + ns) * 16 + r;
            bfr[ns].q[0] = *reinterpret_cast<const v4u*>(sB + col * AST + 8 * hf);
            bfr[ns].q[1] = *reinterpret_cast<const v4u*>(sB + col * AST + 16 + 8 * hf);
        }
        #pragma unroll
        for (int ms = 0; ms < MSUB; ++ms)
            #pragma unroll
            for (int ns = 0; ns < NSUB; ++ns)
                acc[ms][ns] = __builtin_amdgcn_wmma_f32_16x16x32_bf16(
                    false, afr[ms].v, false, bfr[ns].v, (short)0, acc[ms][ns], false, false);
        __syncthreads();
    }

    // epilogue
    #pragma unroll
    for (int ms = 0; ms < MSUB; ++ms) {
        #pragma unroll
        for (int ns = 0; ns < NSUB; ++ns) {
            long ncolg = (long)n0 + (wn * NSUB + ns) * 16 + r;
            #pragma unroll
            for (int j = 0; j < 8; ++j) {
                int m = m0 + (wm * MSUB + ms) * 16 + j + 8 * hf;
                float v = acc[ms][ns][j];
                if (EPI == 0) { if (bias) v += bias[m]; }
                else          { v -= asum[z * KCLU + m] * cent[(long)m * DIMC + ncolg]; }
                C[(long)z * cbstride + (long)m * ldc + ncolg] = v;
            }
        }
    }
}

// ---------------------------------------------------------------------------
__global__ __launch_bounds__(256)
void zero_kernel(float* p, int n)
{
    int i = blockIdx.x * 256 + threadIdx.x;
    if (i < n) p[i] = 0.0f;
}

__global__ __launch_bounds__(256)
void cvt_bf16_kernel(const float* __restrict__ X, __bf16* __restrict__ Y, long n)
{
    long stride = (long)gridDim.x * 256;
    for (long i = (long)blockIdx.x * 256 + threadIdx.x; i < n; i += stride)
        Y[i] = to_bf16(X[i]);
}

__global__ __launch_bounds__(256)
void stats_kernel(const float* __restrict__ X, long N, float* __restrict__ stats, int M)
{
    int m = blockIdx.x;
    float s = 0.f, s2 = 0.f;
    for (long i = threadIdx.x; i < N; i += 256) {
        float v = X[(long)m * N + i];
        s += v; s2 += v * v;
    }
    __shared__ float rs[256], rq[256];
    rs[threadIdx.x] = s; rq[threadIdx.x] = s2;
    __syncthreads();
    for (int o = 128; o > 0; o >>= 1) {
        if (threadIdx.x < o) { rs[threadIdx.x] += rs[threadIdx.x + o]; rq[threadIdx.x] += rq[threadIdx.x + o]; }
        __syncthreads();
    }
    if (threadIdx.x == 0) { stats[m] = rs[0]; stats[M + m] = rq[0]; }
}

__global__ __launch_bounds__(256)
void bnrelu_kernel(const float* __restrict__ X, __bf16* __restrict__ Y,
                   const float* __restrict__ stats, const float* __restrict__ g,
                   const float* __restrict__ be, int M, long N)
{
    long total  = (long)M * N;
    long stride = (long)gridDim.x * 256;
    for (long i = (long)blockIdx.x * 256 + threadIdx.x; i < total; i += stride) {
        int   m    = (int)(i / N);
        float mean = stats[m] / (float)N;
        float var  = stats[M + m] / (float)N - mean * mean;
        float v    = (X[i] - mean) * rsqrtf(var + 1e-5f) * g[m] + be[m];
        Y[i] = to_bf16(fmaxf(v, 0.0f));
    }
}

__global__ __launch_bounds__(256)
void softmax_kernel(const float* __restrict__ L, __bf16* __restrict__ S, float* __restrict__ asum)
{
    long n = (long)blockIdx.x * 256 + threadIdx.x;
    __shared__ float as[KCLU];
    if (threadIdx.x < KCLU) as[threadIdx.x] = 0.0f;
    __syncthreads();
    float p[KCLU];
    float mx = -3.4e38f;
    #pragma unroll
    for (int k = 0; k < KCLU; ++k) { p[k] = L[(long)k * NTOT + n]; mx = fmaxf(mx, p[k]); }
    float s = 0.f;
    #pragma unroll
    for (int k = 0; k < KCLU; ++k) { p[k] = __expf(p[k] - mx); s += p[k]; }
    float inv = 1.0f / s;
    #pragma unroll
    for (int k = 0; k < KCLU; ++k) {
        float v = p[k] * inv;
        S[(long)k * NTOT + n] = to_bf16(v);
        atomicAdd(&as[k], v);
    }
    __syncthreads();
    int b = (int)(((long)blockIdx.x * 256) / NUMP);
    if (threadIdx.x < KCLU) atomicAdd(&asum[b * KCLU + threadIdx.x], as[threadIdx.x]);
}

__global__ __launch_bounds__(256)
void ln_kernel(const float* __restrict__ X, float* __restrict__ Y,
               const float* __restrict__ g, const float* __restrict__ b)
{
    int row = blockIdx.x;            // BS*KCLU rows of DIMC
    const float* x = X + (long)row * DIMC;
    float*       y = Y + (long)row * DIMC;
    float lv[4];
    float s = 0.f;
    #pragma unroll
    for (int i = 0; i < 4; ++i) { lv[i] = x[threadIdx.x + i * 256]; s += lv[i]; }
    __shared__ float red[256];
    red[threadIdx.x] = s; __syncthreads();
    for (int o = 128; o > 0; o >>= 1) { if (threadIdx.x < o) red[threadIdx.x] += red[threadIdx.x + o]; __syncthreads(); }
    float mean = red[0] * (1.0f / DIMC);
    __syncthreads();
    float s2 = 0.f;
    #pragma unroll
    for (int i = 0; i < 4; ++i) { float d = lv[i] - mean; s2 += d * d; }
    red[threadIdx.x] = s2; __syncthreads();
    for (int o = 128; o > 0; o >>= 1) { if (threadIdx.x < o) red[threadIdx.x] += red[threadIdx.x + o]; __syncthreads(); }
    float inv = rsqrtf(red[0] * (1.0f / DIMC) + 1e-5f);
    #pragma unroll
    for (int i = 0; i < 4; ++i) {
        int d = threadIdx.x + i * 256;
        y[d] = (lv[i] - mean) * inv * g[d] + b[d];
    }
}

__global__ __launch_bounds__(256)
void maxpool_kernel(const float* __restrict__ V, float* __restrict__ P)
{
    int idx = blockIdx.x * 256 + threadIdx.x;      // BS*DIMC
    int b = idx >> 10, d = idx & 1023;
    float m = -3.4e38f;
    #pragma unroll
    for (int k = 0; k < KCLU; ++k) m = fmaxf(m, V[((long)b * KCLU + k) * DIMC + d]);
    P[idx] = m;
}

__global__ __launch_bounds__(256)
void writeout_kernel(const float* __restrict__ PO, const float* __restrict__ MO, float* __restrict__ out)
{
    int idx = blockIdx.x * 256 + threadIdx.x;
    if (idx < BSZ * KCLU * PROJD) {                           // out_vlad (b,k,proj) = PO[proj][b*32+k]
        int pch = idx & (PROJD - 1);
        int n   = idx >> 7;
        out[1 + idx] = PO[(long)pch * (BSZ * KCLU) + n];
    } else if (idx < BSZ * KCLU * PROJD + BSZ * PROJD) {      // c_vlad (b,proj) = MO[proj][b]
        int i = idx - BSZ * KCLU * PROJD;
        int b = i >> 7, pch = i & (PROJD - 1);
        out[1 + BSZ * KCLU * PROJD + i] = MO[pch * BSZ + b];
    }
}

__global__ __launch_bounds__(256)
void orth_kernel(const float* __restrict__ cent, float* __restrict__ out)
{
    __shared__ float nrm[KCLU];
    int tid = threadIdx.x;
    if (tid < KCLU) {
        float ss = 0.f;
        for (int d = 0; d < DIMC; ++d) { float v = cent[tid * DIMC + d]; ss += v * v; }
        nrm[tid] = fmaxf(sqrtf(ss), 1e-12f);
    }
    __syncthreads();
    float acc = 0.f;
    for (int pr = tid; pr < KCLU * KCLU; pr += 256) {
        int i = pr >> 5, j = pr & 31;
        float dot = 0.f;
        for (int d = 0; d < DIMC; ++d) dot += cent[i * DIMC + d] * cent[j * DIMC + d];
        float v = dot / (nrm[i] * nrm[j]);
        acc += fabsf(v - (i == j ? 1.0f : 0.0f));
    }
    __shared__ float red[256];
    red[tid] = acc; __syncthreads();
    for (int o = 128; o > 0; o >>= 1) { if (tid < o) red[tid] += red[tid + o]; __syncthreads(); }
    if (tid == 0) out[0] = 0.002f * red[0];
}

// ---------------------------------------------------------------------------
extern "C" void kernel_launch(void* const* d_in, const int* in_sizes, int n_in,
                              void* d_out, int out_size, void* d_ws, size_t ws_size,
                              hipStream_t stream)
{
    // d_in order: features, centroids, then (w1,b1,g1,be1,w2,b2,g2,be2,w3,b3) for 'c','p','m', then ln_g, ln_b
    enum { F = 0, CENT = 1,
           CW1 = 2, CB1, CG1, CBE1, CW2, CB2, CG2, CBE2, CW3, CB3,
           PW1 = 12, PB1, PG1, PBE1, PW2, PB2, PG2, PBE2, PW3, PB3,
           MW1 = 22, MB1, MG1, MBE1, MW2, MB2, MG2, MBE2, MW3, MB3,
           LNG = 32, LNB = 33 };
    const float* in_f[34];
    for (int i = 0; i < 34 && i < n_in; ++i) in_f[i] = (const float*)d_in[i];
    float* out = (float*)d_out;
    (void)in_sizes; (void)out_size; (void)ws_size;

    // workspace carve-up
    size_t off = 0;
    auto take = [&](size_t bytes) -> void* {
        void* p = (char*)d_ws + off;
        off += (bytes + 255) & ~(size_t)255;
        return p;
    };
    float*  H    = (float*)take((size_t)512 * NTOT * 4);           // conv hidden logits (fp32)
    __bf16* Hb   = (__bf16*)take((size_t)512 * NTOT * 2);          // post BN+ReLU (bf16)
    float*  SAL  = (float*)take((size_t)KCLU * NTOT * 4);          // soft-assign logits
    __bf16* SAB  = (__bf16*)take((size_t)KCLU * NTOT * 2);         // softmax (bf16)
    float*  ASUM = (float*)take(BSZ * KCLU * 4);
    float*  STAT = (float*)take(1024 * 4);
    float*  VLR  = (float*)take((size_t)BSZ * KCLU * DIMC * 4);    // vlad raw
    float*  VLN  = (float*)take((size_t)BSZ * KCLU * DIMC * 4);    // vlad layernormed
    float*  POOL = (float*)take((size_t)BSZ * DIMC * 4);
    float*  HP   = (float*)take((size_t)512 * 512 * 4);
    __bf16* HPb  = (__bf16*)take((size_t)512 * 512 * 2);
    float*  PO   = (float*)take((size_t)PROJD * 512 * 4);
    float*  HM   = (float*)take((size_t)512 * BSZ * 4);
    __bf16* HMb  = (__bf16*)take((size_t)512 * BSZ * 2);
    float*  MO   = (float*)take((size_t)PROJD * BSZ * 4);
    // bf16 copies of weight matrices (A operands -> raw-byte TDM staging)
    __bf16* WCB1 = (__bf16*)take((size_t)512 * 1024 * 2);
    __bf16* WCB2 = (__bf16*)take((size_t)512 * 512 * 2);
    __bf16* WCB3 = (__bf16*)take((size_t)KCLU * 512 * 2);
    __bf16* WPB1 = (__bf16*)take((size_t)512 * 1024 * 2);
    __bf16* WPB2 = (__bf16*)take((size_t)512 * 512 * 2);
    __bf16* WPB3 = (__bf16*)take((size_t)PROJD * 512 * 2);
    __bf16* WMB1 = (__bf16*)take((size_t)512 * 1024 * 2);
    __bf16* WMB2 = (__bf16*)take((size_t)512 * 512 * 2);
    __bf16* WMB3 = (__bf16*)take((size_t)PROJD * 512 * 2);

    const long FB = (long)DIMC * NUMP;   // per-batch feature stride
    dim3 blk(256);

    zero_kernel<<<dim3(2), blk, 0, stream>>>(ASUM, BSZ * KCLU);

    // one-time weight conversions fp32 -> bf16
    cvt_bf16_kernel<<<dim3(512), blk, 0, stream>>>(in_f[CW1], WCB1, (long)512 * 1024);
    cvt_bf16_kernel<<<dim3(256), blk, 0, stream>>>(in_f[CW2], WCB2, (long)512 * 512);
    cvt_bf16_kernel<<<dim3(64),  blk, 0, stream>>>(in_f[CW3], WCB3, (long)KCLU * 512);
    cvt_bf16_kernel<<<dim3(512), blk, 0, stream>>>(in_f[PW1], WPB1, (long)512 * 1024);
    cvt_bf16_kernel<<<dim3(256), blk, 0, stream>>>(in_f[PW2], WPB2, (long)512 * 512);
    cvt_bf16_kernel<<<dim3(128), blk, 0, stream>>>(in_f[PW3], WPB3, (long)PROJD * 512);
    cvt_bf16_kernel<<<dim3(512), blk, 0, stream>>>(in_f[MW1], WMB1, (long)512 * 1024);
    cvt_bf16_kernel<<<dim3(256), blk, 0, stream>>>(in_f[MW2], WMB2, (long)512 * 512);
    cvt_bf16_kernel<<<dim3(128), blk, 0, stream>>>(in_f[MW3], WMB3, (long)PROJD * 512);

    // ---- soft-assign conv: 1024 -> 512 -> 512 -> 32 over 65536 positions ----
    gemm_wmma<128,128,2,4,float,false,0><<<dim3(NTOT/128, 4, 1), blk, 0, stream>>>(
        WCB1, 1024, 0, in_f[F], NUMP, 0, NUMP, FB,
        H, NTOT, 0, in_f[CB1], nullptr, nullptr, 1024);
    stats_kernel<<<dim3(512), blk, 0, stream>>>(H, NTOT, STAT, 512);
    bnrelu_kernel<<<dim3(8192), blk, 0, stream>>>(H, Hb, STAT, in_f[CG1], in_f[CBE1], 512, NTOT);

    gemm_wmma<128,128,2,4,__bf16,false,0><<<dim3(NTOT/128, 4, 1), blk, 0, stream>>>(
        WCB2, 512, 0, Hb, NTOT, 0, 0, 0,
        H, NTOT, 0, in_f[CB2], nullptr, nullptr, 512);
    stats_kernel<<<dim3(512), blk, 0, stream>>>(H, NTOT, STAT, 512);
    bnrelu_kernel<<<dim3(8192), blk, 0, stream>>>(H, Hb, STAT, in_f[CG2], in_f[CBE2], 512, NTOT);

    gemm_wmma<32,256,1,8,__bf16,false,0><<<dim3(NTOT/256, 1, 1), blk, 0, stream>>>(
        WCB3, 512, 0, Hb, NTOT, 0, 0, 0,
        SAL, NTOT, 0, in_f[CB3], nullptr, nullptr, 512);

    softmax_kernel<<<dim3(NTOT/256), blk, 0, stream>>>(SAL, SAB, ASUM);

    // ---- fused VLAD: per batch C(32x1024) = sa(32x4096) x f^T(4096x1024), minus a_sum*centroids
    gemm_wmma<32,256,1,8,float,true,1><<<dim3(DIMC/256, 1, BSZ), blk, 0, stream>>>(
        SAB, NTOT, NUMP, in_f[F], NUMP, FB, 0, 0,
        VLR, DIMC, (long)KCLU * DIMC, nullptr, ASUM, in_f[CENT], NUMP);

    ln_kernel<<<dim3(BSZ * KCLU), blk, 0, stream>>>(VLR, VLN, in_f[LNG], in_f[LNB]);
    maxpool_kernel<<<dim3((BSZ * DIMC) / 256), blk, 0, stream>>>(VLN, POOL);

    // ---- projector conv on vlad_t: 1024 -> 512 -> 512 -> 128 over 512 positions ----
    gemm_wmma<128,128,2,4,float,true,0><<<dim3(4, 4, 1), blk, 0, stream>>>(
        WPB1, 1024, 0, VLN, DIMC, 0, 0, 0,
        HP, 512, 0, in_f[PB1], nullptr, nullptr, 1024);
    stats_kernel<<<dim3(512), blk, 0, stream>>>(HP, 512, STAT, 512);
    bnrelu_kernel<<<dim3(1024), blk, 0, stream>>>(HP, HPb, STAT, in_f[PG1], in_f[PBE1], 512, 512);

    gemm_wmma<128,128,2,4,__bf16,false,0><<<dim3(4, 4, 1), blk, 0, stream>>>(
        WPB2, 512, 0, HPb, 512, 0, 0, 0,
        HP, 512, 0, in_f[PB2], nullptr, nullptr, 512);
    stats_kernel<<<dim3(512), blk, 0, stream>>>(HP, 512, STAT, 512);
    bnrelu_kernel<<<dim3(1024), blk, 0, stream>>>(HP, HPb, STAT, in_f[PG2], in_f[PBE2], 512, 512);

    gemm_wmma<128,128,2,4,__bf16,false,0><<<dim3(4, 1, 1), blk, 0, stream>>>(
        WPB3, 512, 0, HPb, 512, 0, 0, 0,
        PO, 512, 0, in_f[PB3], nullptr, nullptr, 512);

    // ---- predictor MLP on pooled: 1024 -> 512 -> 512 -> 128 over 16 rows ----
    gemm_wmma<128,16,8,1,float,true,0><<<dim3(1, 4, 1), blk, 0, stream>>>(
        WMB1, 1024, 0, POOL, DIMC, 0, 0, 0,
        HM, BSZ, 0, in_f[MB1], nullptr, nullptr, 1024);
    stats_kernel<<<dim3(512), blk, 0, stream>>>(HM, BSZ, STAT, 512);
    bnrelu_kernel<<<dim3(32), blk, 0, stream>>>(HM, HMb, STAT, in_f[MG1], in_f[MBE1], 512, BSZ);

    gemm_wmma<128,16,8,1,__bf16,false,0><<<dim3(1, 4, 1), blk, 0, stream>>>(
        WMB2, 512, 0, HMb, BSZ, 0, 0, 0,
        HM, BSZ, 0, in_f[MB2], nullptr, nullptr, 512);
    stats_kernel<<<dim3(512), blk, 0, stream>>>(HM, BSZ, STAT, 512);
    bnrelu_kernel<<<dim3(32), blk, 0, stream>>>(HM, HMb, STAT, in_f[MG2], in_f[MBE2], 512, BSZ);

    gemm_wmma<128,16,8,1,__bf16,false,0><<<dim3(1, 1, 1), blk, 0, stream>>>(
        WMB3, 512, 0, HMb, BSZ, 0, 0, 0,
        MO, BSZ, 0, in_f[MB3], nullptr, nullptr, 512);

    // ---- outputs ----
    writeout_kernel<<<dim3((BSZ*KCLU*PROJD + BSZ*PROJD + 255) / 256), blk, 0, stream>>>(PO, MO, out);
    orth_kernel<<<dim3(1), blk, 0, stream>>>(in_f[CENT], out);
}
